// Hyena_Y_22797686407469
// MI455X (gfx1250) — matrix operand order repaired and checked by reading, not compiled, not run on hardware
//
#include <hip/hip_runtime.h>
#include <hip/hip_bf16.h>

// ---------------------------------------------------------------------------
// Hyena-style operator for MI455X (gfx1250, wave32, WMMA).
//   uc = u @ w_in^T + b_in ; q,k,v = split(uc) ; kv = k*v
//   x  = causal_depthwise_conv4(kv) + b_filt ; y = q*x
//   out = y @ w_out^T + b_out
// GEMMs run on V_WMMA_F32_16X16X32_BF16 (f32 accumulate). Tiles are staged
// into LDS with GLOBAL_LOAD_ASYNC_TO_LDS_B128 (ASYNCcnt), double-buffered so
// copies for tile k+1 overlap WMMAs on tile k.
// ---------------------------------------------------------------------------

typedef __attribute__((ext_vector_type(16))) __bf16 bf16x16;
typedef __attribute__((ext_vector_type(8)))  __bf16 bf16x8;
typedef __attribute__((ext_vector_type(4)))  __bf16 bf16x4;
typedef __attribute__((ext_vector_type(8)))  float  f32x8;
typedef __attribute__((ext_vector_type(4)))  float  f32x4;

// GCC-style int4 to match the async-copy builtin's pointee type exactly.
typedef int v4i_g __attribute__((vector_size(16)));
typedef __attribute__((address_space(1))) v4i_g* as1_v4i;
typedef __attribute__((address_space(3))) v4i_g* as3_v4i;

#define BSZ 2
#define LSZ 4096
#define DSZ 2048
#define MSZ (BSZ*LSZ)          // 8192 rows (B*L)
#define BK  32                 // K per pipeline stage (one WMMA K-step)
#define LDK 40                 // LDS row stride in elems (80B, 16B-aligned)

#if defined(__has_builtin)
#if __has_builtin(__builtin_amdgcn_global_load_async_to_lds_b128)
#define HAS_ASYNC_LDS 1
#endif
#endif

// 16B global -> LDS copy. Async DMA path (ASYNCcnt) when available, else a
// plain load+ds_store pair.
__device__ __forceinline__ void cp_b128(void* lds, const void* g) {
#ifdef HAS_ASYNC_LDS
  __builtin_amdgcn_global_load_async_to_lds_b128(
      (as1_v4i)(unsigned long long)g,
      (as3_v4i)(unsigned)(unsigned long long)lds,
      0, 0);
#else
  *(bf16x8*)lds = *(const bf16x8*)g;
#endif
}

__device__ __forceinline__ void cp_wait() {
#ifdef HAS_ASYNC_LDS
#if __has_builtin(__builtin_amdgcn_s_wait_asynccnt)
  __builtin_amdgcn_s_wait_asynccnt(0);
#else
  asm volatile("s_wait_asynccnt 0x0" ::: "memory");
#endif
#endif
}

__device__ __forceinline__ bf16x16 join8(bf16x8 lo, bf16x8 hi) {
  bf16x16 r;
#pragma unroll
  for (int i = 0; i < 8; ++i) { r[i] = lo[i]; r[i + 8] = hi[i]; }
  return r;
}

// ---------------------------------------------------------------------------
// Kernel 1: convert w_in, w_out and u fp32 -> bf16 (one float4 per thread;
// total chunks divide the grid exactly).
// ---------------------------------------------------------------------------
__global__ __launch_bounds__(256) void hyena_cvt(
    const float* __restrict__ w_in, const float* __restrict__ w_out,
    const float* __restrict__ u,
    __bf16* __restrict__ w_in_bf, __bf16* __restrict__ w_out_bf,
    __bf16* __restrict__ u_bf) {
  const int nA = 3 * DSZ * DSZ / 4;   // 3,145,728
  const int nB = DSZ * DSZ / 4;       // 1,048,576
  int idx = blockIdx.x * 256 + threadIdx.x;
  const float* src; __bf16* dst; int j;
  if (idx < nA)            { src = w_in;  dst = w_in_bf;  j = idx; }
  else if (idx < nA + nB)  { src = w_out; dst = w_out_bf; j = idx - nA; }
  else                     { src = u;     dst = u_bf;     j = idx - nA - nB; }
  f32x4 v = ((const f32x4*)src)[j];
  bf16x4 o;
#pragma unroll
  for (int i = 0; i < 4; ++i) o[i] = (__bf16)v[i];
  ((bf16x4*)dst)[j] = o;
}

// ---------------------------------------------------------------------------
// Kernel 2: fused QKV projection. Workgroup tile 128(M) x 64(D); computes
// q, k, v together (shared A tile). 8 waves in 4(m) x 2(n); each wave owns a
// 32x32 patch per matrix = 2x2 WMMA tiles. Double-buffered async pipeline.
// Epilogue: q -> bf16, kv = (k+bk)*(v+bv) -> f32.
// ---------------------------------------------------------------------------
__global__ __launch_bounds__(256) void hyena_gemm_qkv(
    const __bf16* __restrict__ u_bf, const __bf16* __restrict__ w_in_bf,
    const float* __restrict__ b_in,
    __bf16* __restrict__ q_bf, float* __restrict__ kv) {
  __shared__ __bf16 As[2][128][LDK];      // 20,480 B
  __shared__ __bf16 Bs[2][3][64][LDK];    // 30,720 B

  const int tid  = threadIdx.x;
  const int lane = tid & 31;
  const int wave = tid >> 5;
  const int wm   = wave & 3;
  const int wn   = wave >> 2;
  const int m0   = blockIdx.x * 128;
  const int n0   = blockIdx.y * 64;
  const int r16  = lane & 15;
  const int hi   = lane >> 4;

  f32x8 acc[3][2][2] = {};                // [q/k/v][sub_m][sub_n]

  // per-thread staging assignments (16B chunks, BK=32 -> 4 chunks/row)
  const int a_row0 = tid >> 2, a_c = (tid & 3) * 8;
  const int NT = DSZ / BK;                // 64 pipeline stages

  auto stage = [&](int kt, int b) {
#pragma unroll
    for (int p = 0; p < 2; ++p) {         // A: 128 rows x 4 chunks = 512
      int row = a_row0 + p * 64;
      cp_b128(&As[b][row][a_c],
              u_bf + (size_t)(m0 + row) * DSZ + kt + a_c);
    }
#pragma unroll
    for (int p = 0; p < 3; ++p) {         // B: 3 x 64 rows x 4 chunks = 768
      int idx = tid + p * 256;            // 0..767
      int mat = idx >> 8, rem = idx & 255;
      int row = rem >> 2, c = (rem & 3) * 8;
      cp_b128(&Bs[b][mat][row][c],
              w_in_bf + (size_t)(mat * DSZ + n0 + row) * DSZ + kt + c);
    }
  };

  stage(0, 0);
  for (int it = 0; it < NT; ++it) {
    const int cur = it & 1;
    cp_wait();              // my copies (incl. those for tile `it`) complete
    __syncthreads();        // everyone's copies done; prev compute finished
    if (it + 1 < NT) stage((it + 1) * BK, cur ^ 1);

    bf16x16 afrag[2];
#pragma unroll
    for (int sm = 0; sm < 2; ++sm) {
      const __bf16* ap = &As[cur][wm * 32 + sm * 16 + r16][hi * 8];
      afrag[sm] = join8(*(const bf16x8*)ap, *(const bf16x8*)(ap + 16));
    }
#pragma unroll
    for (int mat = 0; mat < 3; ++mat) {
#pragma unroll
      for (int sn = 0; sn < 2; ++sn) {
        const __bf16* bp = &Bs[cur][mat][wn * 32 + sn * 16 + r16][hi * 8];
        bf16x16 bfrag = join8(*(const bf16x8*)bp, *(const bf16x8*)(bp + 16));
#pragma unroll
        for (int sm = 0; sm < 2; ++sm) {
          acc[mat][sm][sn] = __builtin_amdgcn_wmma_f32_16x16x32_bf16(
              false, afrag[sm], false, bfrag, (short)0, acc[mat][sm][sn],
              false, false);
        }
      }
    }
  }

  // ---- epilogue ----
#pragma unroll
  for (int sm = 0; sm < 2; ++sm) {
#pragma unroll
    for (int sn = 0; sn < 2; ++sn) {
      const int col  = n0 + wn * 32 + sn * 16 + r16;
      const float bq = b_in[col];
      const float bk = b_in[DSZ + col];
      const float bv = b_in[2 * DSZ + col];
      const int rowb = m0 + wm * 32 + sm * 16 + hi * 8;   // C: M = e + 8*hi
#pragma unroll
      for (int e = 0; e < 8; ++e) {
        size_t off = (size_t)(rowb + e) * DSZ + col;
        float qf = acc[0][sm][sn][e] + bq;
        float kf = acc[1][sm][sn][e] + bk;
        float vf = acc[2][sm][sn][e] + bv;
        q_bf[off] = (__bf16)qf;
        kv[off]   = kf * vf;
      }
    }
  }
}

// ---------------------------------------------------------------------------
// Kernel 3: causal 4-tap depthwise conv along L + y = q * x, output bf16.
// ---------------------------------------------------------------------------
__global__ __launch_bounds__(256) void hyena_conv_gate(
    const __bf16* __restrict__ q_bf, const float* __restrict__ kv,
    const float* __restrict__ w_filt, const float* __restrict__ b_filt,
    __bf16* __restrict__ y_bf) {
  size_t idx = (size_t)blockIdx.x * 256 + threadIdx.x;  // < B*L*D
  int d = (int)(idx & (DSZ - 1));
  int l = (int)((idx >> 11) & (LSZ - 1));
  const float* wf = w_filt + d * 4;                     // w_filt[d][j]
  float acc = b_filt[d];
#pragma unroll
  for (int j = 0; j < 4; ++j) {
    int dl = l - 3 + j;                                 // tap position
    if (dl >= 0) {
      long long off = (long long)idx + (long long)(j - 3) * DSZ;
      acc += kv[off] * wf[j];
    }
  }
  float y = (float)q_bf[idx] * acc;
  y_bf[idx] = (__bf16)y;
}

// ---------------------------------------------------------------------------
// Kernel 4: out = y_bf @ w_out_bf^T + b_out  (f32 output). Same pipeline,
// single B matrix.
// ---------------------------------------------------------------------------
__global__ __launch_bounds__(256) void hyena_gemm_out(
    const __bf16* __restrict__ y_bf, const __bf16* __restrict__ w_out_bf,
    const float* __restrict__ b_out, float* __restrict__ out) {
  __shared__ __bf16 As[2][128][LDK];      // 20,480 B
  __shared__ __bf16 Bs[2][64][LDK];       // 10,240 B

  const int tid  = threadIdx.x;
  const int lane = tid & 31;
  const int wave = tid >> 5;
  const int wm   = wave & 3;
  const int wn   = wave >> 2;
  const int m0   = blockIdx.x * 128;
  const int n0   = blockIdx.y * 64;
  const int r16  = lane & 15;
  const int hi   = lane >> 4;

  f32x8 acc[2][2] = {};

  const int a_row0 = tid >> 2, a_c = (tid & 3) * 8;
  const int NT = DSZ / BK;

  auto stage = [&](int kt, int b) {
#pragma unroll
    for (int p = 0; p < 2; ++p) {         // A: 512 chunks
      int row = a_row0 + p * 64;
      cp_b128(&As[b][row][a_c],
              y_bf + (size_t)(m0 + row) * DSZ + kt + a_c);
    }
    {                                     // B: 256 chunks
      int row = tid >> 2, c = (tid & 3) * 8;
      cp_b128(&Bs[b][row][c],
              w_out_bf + (size_t)(n0 + row) * DSZ + kt + c);
    }
  };

  stage(0, 0);
  for (int it = 0; it < NT; ++it) {
    const int cur = it & 1;
    cp_wait();
    __syncthreads();
    if (it + 1 < NT) stage((it + 1) * BK, cur ^ 1);

    bf16x16 afrag[2];
#pragma unroll
    for (int sm = 0; sm < 2; ++sm) {
      const __bf16* ap = &As[cur][wm * 32 + sm * 16 + r16][hi * 8];
      afrag[sm] = join8(*(const bf16x8*)ap, *(const bf16x8*)(ap + 16));
    }
#pragma unroll
    for (int sn = 0; sn < 2; ++sn) {
      const __bf16* bp = &Bs[cur][wn * 32 + sn * 16 + r16][hi * 8];
      bf16x16 bfrag = join8(*(const bf16x8*)bp, *(const bf16x8*)(bp + 16));
#pragma unroll
      for (int sm = 0; sm < 2; ++sm) {
        acc[sm][sn] = __builtin_amdgcn_wmma_f32_16x16x32_bf16(
            false, afrag[sm], false, bfrag, (short)0, acc[sm][sn],
            false, false);
      }
    }
  }

#pragma unroll
  for (int sm = 0; sm < 2; ++sm) {
#pragma unroll
    for (int sn = 0; sn < 2; ++sn) {
      const int col  = n0 + wn * 32 + sn * 16 + r16;
      const float bb = b_out[col];
      const int rowb = m0 + wm * 32 + sm * 16 + hi * 8;
#pragma unroll
      for (int e = 0; e < 8; ++e) {
        out[(size_t)(rowb + e) * DSZ + col] = acc[sm][sn][e] + bb;
      }
    }
  }
}

// ---------------------------------------------------------------------------
// Launch. Workspace layout (bytes):
//   [0,            25165824)  w_in_bf   (3D x D bf16)
//   [25165824,     33554432)  w_out_bf  (D x D bf16)
//   [33554432,     67108864)  u_bf      (B*L x D bf16)
//   [67108864,    100663296)  q_bf      (B*L x D bf16)
//   [100663296,   167772160)  kv        (B*L x D f32)
//   [167772160,   201326592)  y_bf      (B*L x D bf16)
// Total: 192 MiB.
// ---------------------------------------------------------------------------
extern "C" void kernel_launch(void* const* d_in, const int* in_sizes, int n_in,
                              void* d_out, int out_size, void* d_ws, size_t ws_size,
                              hipStream_t stream) {
  const float* u      = (const float*)d_in[0];
  const float* w_in   = (const float*)d_in[1];
  const float* b_in   = (const float*)d_in[2];
  const float* w_filt = (const float*)d_in[3];
  const float* b_filt = (const float*)d_in[4];
  const float* w_out  = (const float*)d_in[5];
  const float* b_out  = (const float*)d_in[6];
  float* out = (float*)d_out;

  char* ws = (char*)d_ws;
  __bf16* w_in_bf  = (__bf16*)(ws);
  __bf16* w_out_bf = (__bf16*)(ws + 25165824ULL);
  __bf16* u_bf     = (__bf16*)(ws + 33554432ULL);
  __bf16* q_bf     = (__bf16*)(ws + 67108864ULL);
  float*  kv       = (float*) (ws + 100663296ULL);
  __bf16* y_bf     = (__bf16*)(ws + 167772160ULL);

  // 1) fp32 -> bf16 conversions: (3D*D + D*D + M*D)/4 = 8,388,608 float4s
  hipLaunchKernelGGL(hyena_cvt, dim3(32768), dim3(256), 0, stream,
                     w_in, w_out, u, w_in_bf, w_out_bf, u_bf);
  // 2) fused QKV projection: grid (M/128, D/64)
  hipLaunchKernelGGL(hyena_gemm_qkv, dim3(MSZ / 128, DSZ / 64), dim3(256), 0,
                     stream, u_bf, w_in_bf, b_in, q_bf, kv);
  // 3) causal conv + gate
  hipLaunchKernelGGL(hyena_conv_gate, dim3((BSZ * LSZ * DSZ) / 256), dim3(256),
                     0, stream, q_bf, kv, w_filt, b_filt, y_bf);
  // 4) output projection
  hipLaunchKernelGGL(hyena_gemm_out, dim3(MSZ / 128, DSZ / 64), dim3(256), 0,
                     stream, y_bf, w_out_bf, b_out, out);
}
